// DiagonalSSM_62302795596621
// MI455X (gfx1250) — matrix operand order repaired and checked by reading, not compiled, not run on hardware
//
#include <hip/hip_runtime.h>
#include <math.h>

// ---------------------------------------------------------------------------
// Model dims (fixed by the reference)
// B=4 S=2048 D=1024 H=8 N=16 HN=128 L=2 DFF=4096 C=1024, M = B*S = 8192
// All GEMM dims are multiples of 256 (M), 128 (N) and 32 (K).
// ---------------------------------------------------------------------------

typedef _Float16 half_t;
typedef __attribute__((ext_vector_type(16))) half_t v16h;
typedef __attribute__((ext_vector_type(8)))  half_t v8h;
typedef __attribute__((ext_vector_type(8)))  float  v8f;

static __device__ __forceinline__ float sigmoidf_(float x) {
    return 1.0f / (1.0f + expf(-x));
}

// ---------------------------------------------------------------------------
// WMMA GEMM:  C[M,N] = A[M,K] (f16, row-major) x Bt[N,K]^T (f16, N-major) + epi
// Block tile 256(M) x 128(N), BK=32, 256 threads = 8 waves in a 4(M)x2(N) grid.
// Each wave: 64x64 = 4x4 subtiles of 16x16 -> 16 WMMAs per BK step.
// Double-buffered LDS + register staging: one barrier per K-step, global loads
// for tile i+1 are in flight while tile i's 16 WMMAs execute.
// EPI: 0 = bias -> f32          1 = bias + residual -> f32 (in-place h update)
//      2 = silu(acc+bias) -> f16          3 = aux * (acc+bias) -> f16 (SwiGLU)
//      4 = exact gelu(acc+bias) -> f16
// ---------------------------------------------------------------------------
constexpr int BM = 256, BN = 128, BK = 32;
constexpr int LDT = BK + 8;  // padded LDS stride (halfs); 80B keeps 16B align

template <int EPI>
__global__ __launch_bounds__(256) void gemm_f16_k(
    const half_t* __restrict__ A,
    const half_t* __restrict__ Bt,
    const float*  __restrict__ bias,
    const float*  res,              // no restrict: may alias Cf (in-place)
    const half_t* __restrict__ aux,
    float*        Cf,               // no restrict: may alias res
    half_t*       __restrict__ Cb,
    int M, int N, int K)
{
    __shared__ half_t sA[2][BM * LDT];
    __shared__ half_t sB[2][BN * LDT];

    const int tid  = threadIdx.x;
    const int lane = tid & 31;
    const int wave = tid >> 5;
    const int wm   = wave >> 1;   // 0..3 -> 64-row slab
    const int wn   = wave & 1;    // 0..1 -> 64-col slab
    const int bm   = blockIdx.y * BM;
    const int bn   = blockIdx.x * BN;
    const int ml   = lane & 15;   // M (A) / N (B/C) index within 16x16 subtile
    const int hf   = lane >> 4;   // lane-half

    v8f acc[4][4];
#pragma unroll
    for (int i = 0; i < 4; ++i)
#pragma unroll
        for (int j = 0; j < 4; ++j)
            acc[i][j] = (v8f){0.f, 0.f, 0.f, 0.f, 0.f, 0.f, 0.f, 0.f};

    // staging registers: A tile = 1024 segs of 8 halfs (4/thread), B = 512 (2/thread)
    v8h ra[4], rb[2];

    auto gload = [&](int k0) {
#pragma unroll
        for (int p = 0; p < 4; ++p) {
            const int seg = tid + p * 256;
            const int r = seg >> 2, c = (seg & 3) * 8;
            ra[p] = *(const v8h*)&A[(size_t)(bm + r) * K + k0 + c];
        }
#pragma unroll
        for (int p = 0; p < 2; ++p) {
            const int seg = tid + p * 256;
            const int r = seg >> 2, c = (seg & 3) * 8;
            rb[p] = *(const v8h*)&Bt[(size_t)(bn + r) * K + k0 + c];
        }
    };
    auto lstore = [&](int buf) {
#pragma unroll
        for (int p = 0; p < 4; ++p) {
            const int seg = tid + p * 256;
            const int r = seg >> 2, c = (seg & 3) * 8;
            *(v8h*)&sA[buf][r * LDT + c] = ra[p];
        }
#pragma unroll
        for (int p = 0; p < 2; ++p) {
            const int seg = tid + p * 256;
            const int r = seg >> 2, c = (seg & 3) * 8;
            *(v8h*)&sB[buf][r * LDT + c] = rb[p];
        }
    };

    gload(0);
    lstore(0);
    __syncthreads();

    int buf = 0;
    for (int k0 = 0; k0 < K; k0 += BK) {
        const bool more = (k0 + BK) < K;
        if (more) gload(k0 + BK);  // in flight during the WMMAs below
        if (more) {
            const int r = tid >> 2, c = (tid & 3) * 8;
            __builtin_prefetch(&A[(size_t)(bm + r) * K + k0 + 2 * BK + c], 0, 3);
            __builtin_prefetch(&Bt[(size_t)(bn + r) * K + k0 + 2 * BK + c], 0, 3);
        }

        // --- B fragments (ISA 7.12.2): lane ml=N; K slots {hf*8+0..7, 16+hf*8+0..7}
        v16h bfrag[4];
#pragma unroll
        for (int ni = 0; ni < 4; ++ni) {
            const int col = wn * 64 + ni * 16 + ml;
            v8h lo = *(const v8h*)&sB[buf][col * LDT + hf * 8];
            v8h hi = *(const v8h*)&sB[buf][col * LDT + 16 + hf * 8];
            bfrag[ni] = __builtin_shufflevector(lo, hi, 0, 1, 2, 3, 4, 5, 6, 7,
                                                8, 9, 10, 11, 12, 13, 14, 15);
        }
        // --- A fragment per mi: lane ml=M; K slots {hf*8+0..7, 16+hf*8+0..7}
#pragma unroll
        for (int mi = 0; mi < 4; ++mi) {
            const int row = wm * 64 + mi * 16 + ml;
            v8h lo = *(const v8h*)&sA[buf][row * LDT + hf * 8];
            v8h hi = *(const v8h*)&sA[buf][row * LDT + 16 + hf * 8];
            v16h a = __builtin_shufflevector(lo, hi, 0, 1, 2, 3, 4, 5, 6, 7,
                                             8, 9, 10, 11, 12, 13, 14, 15);
#pragma unroll
            for (int ni = 0; ni < 4; ++ni)
                acc[mi][ni] = __builtin_amdgcn_wmma_f32_16x16x32_f16(
                    false, a, false, bfrag[ni], (short)0, acc[mi][ni],
                    false, false);
        }

        if (more) lstore(buf ^ 1);  // waits on loadcnt here, after the WMMAs
        __syncthreads();
        buf ^= 1;
    }

    // --- epilogue: D layout = lane ml=N, VGPR r -> M = hf*8 + r
#pragma unroll
    for (int mi = 0; mi < 4; ++mi) {
#pragma unroll
        for (int ni = 0; ni < 4; ++ni) {
            const int   gn = bn + wn * 64 + ni * 16 + ml;
            const float bv = bias[gn];
#pragma unroll
            for (int r = 0; r < 8; ++r) {
                const int    gm  = bm + wm * 64 + mi * 16 + hf * 8 + r;
                const size_t idx = (size_t)gm * N + gn;
                const float  v   = acc[mi][ni][r] + bv;
                if constexpr (EPI == 0) {
                    Cf[idx] = v;
                } else if constexpr (EPI == 1) {
                    Cf[idx] = res[idx] + v;
                } else if constexpr (EPI == 2) {
                    Cb[idx] = (half_t)(v * sigmoidf_(v));
                } else if constexpr (EPI == 3) {
                    Cb[idx] = (half_t)((float)aux[idx] * v);
                } else {
                    Cb[idx] = (half_t)(0.5f * v * (1.0f + erff(v * 0.70710678118654752f)));
                }
            }
        }
    }
}

// ---------------------------------------------------------------------------
// LayerNorm over D=1024 (one row per block), f32 in -> f16 out
// ---------------------------------------------------------------------------
__global__ __launch_bounds__(256) void ln_k(const float* __restrict__ x,
                                            const float* __restrict__ g,
                                            const float* __restrict__ be,
                                            half_t* __restrict__ out, int Dm)
{
    const int    row = blockIdx.x;
    const int    tid = threadIdx.x;
    const float* xr  = x + (size_t)row * Dm;
    __shared__ float red[256];

    float v[4], s = 0.f;
#pragma unroll
    for (int i = 0; i < 4; ++i) { v[i] = xr[tid + i * 256]; s += v[i]; }
    red[tid] = s;
    __syncthreads();
    for (int off = 128; off > 0; off >>= 1) {
        if (tid < off) red[tid] += red[tid + off];
        __syncthreads();
    }
    const float mean = red[0] * (1.0f / 1024.0f);
    __syncthreads();

    float q = 0.f;
#pragma unroll
    for (int i = 0; i < 4; ++i) { const float d = v[i] - mean; q += d * d; }
    red[tid] = q;
    __syncthreads();
    for (int off = 128; off > 0; off >>= 1) {
        if (tid < off) red[tid] += red[tid + off];
        __syncthreads();
    }
    const float inv = 1.0f / sqrtf(red[0] * (1.0f / 1024.0f) + 1e-5f);

#pragma unroll
    for (int i = 0; i < 4; ++i) {
        const int d = tid + i * 256;
        out[(size_t)row * Dm + d] = (half_t)((v[i] - mean) * inv * g[d] + be[d]);
    }
}

// ---------------------------------------------------------------------------
// Embedding + positional add -> h (f32)
// ---------------------------------------------------------------------------
__global__ void embed_k(const int* __restrict__ x, const float* __restrict__ emb,
                        const float* __restrict__ pos, float* __restrict__ h,
                        int rows, int Ssz, int Dm)
{
    const size_t i = (size_t)blockIdx.x * blockDim.x + threadIdx.x;
    if (i >= (size_t)rows * Dm) return;
    const int row = (int)(i / Dm), d = (int)(i % Dm);
    const int tok = x[row], s = row % Ssz;
    h[i] = emb[(size_t)tok * Dm + d] + pos[(size_t)s * Dm + d];
}

// ---------------------------------------------------------------------------
// Weight convert + transpose: W[K,N] f32 -> Wt[N,K] f16
// ---------------------------------------------------------------------------
__global__ void convtr_k(const float* __restrict__ W, half_t* __restrict__ Wt,
                         int K, int N)
{
    const size_t i = (size_t)blockIdx.x * blockDim.x + threadIdx.x;
    if (i >= (size_t)K * N) return;
    const int k = (int)(i / N), n = (int)(i % N);
    Wt[(size_t)n * K + k] = (half_t)W[i];
}

// ---------------------------------------------------------------------------
// Diagonal SSM scan: 512 independent channels, sigmoid fused; f16 out
// ---------------------------------------------------------------------------
__global__ void scan_k(const float* __restrict__ al, const float* __restrict__ bl,
                       half_t* __restrict__ hs, int Bsz, int Ssz, int HNm)
{
    const int ch = blockIdx.x * blockDim.x + threadIdx.x;
    if (ch >= Bsz * HNm) return;
    const int b = ch / HNm, c = ch % HNm;
    float h = 0.f;
    for (int s = 0; s < Ssz; ++s) {
        const size_t idx = ((size_t)(b * Ssz + s)) * HNm + c;
        const float  a   = sigmoidf_(al[idx]);
        h = a * h + bl[idx];
        hs[idx] = (half_t)h;
    }
}

// ---------------------------------------------------------------------------
extern "C" void kernel_launch(void* const* d_in, const int* in_sizes, int n_in,
                              void* d_out, int out_size, void* d_ws, size_t ws_size,
                              hipStream_t stream)
{
    (void)in_sizes; (void)n_in; (void)out_size; (void)ws_size;
    const int Bsz = 4, Ssz = 2048, Dm = 1024, HNm = 128, DFFm = 4096, Cm = 1024;
    const int Mm  = Bsz * Ssz;  // 8192

    const int*   x      = (const int*)  d_in[0];
    const float* emb    = (const float*)d_in[1];
    const float* pos    = (const float*)d_in[2];
    const float* alphaW = (const float*)d_in[3];
    const float* alphaB = (const float*)d_in[4];
    const float* inW    = (const float*)d_in[5];
    const float* inB    = (const float*)d_in[6];
    const float* outW   = (const float*)d_in[7];
    const float* outB   = (const float*)d_in[8];
    const float* ssmG   = (const float*)d_in[9];
    const float* ssmBe  = (const float*)d_in[10];
    const float* w1W    = (const float*)d_in[11];
    const float* w1B    = (const float*)d_in[12];
    const float* w2W    = (const float*)d_in[13];
    const float* w2B    = (const float*)d_in[14];
    const float* w3W    = (const float*)d_in[15];
    const float* w3B    = (const float*)d_in[16];
    const float* ffG    = (const float*)d_in[17];
    const float* ffBe   = (const float*)d_in[18];
    const float* onG    = (const float*)d_in[19];
    const float* onBe   = (const float*)d_in[20];
    const float* h1W    = (const float*)d_in[21];
    const float* h1B    = (const float*)d_in[22];
    const float* h2W    = (const float*)d_in[23];
    const float* h2B    = (const float*)d_in[24];
    float*       outp   = (float*)d_out;

    // ---- workspace bump allocator
    char*  ws  = (char*)d_ws;
    size_t off = 0;
    auto alloc = [&](size_t bytes) -> char* {
        char* p = ws + off;
        off += (bytes + 255) & ~(size_t)255;
        return p;
    };
    float*  h    = (float*) alloc((size_t)Mm * Dm * 4);
    half_t* xnb  = (half_t*)alloc((size_t)Mm * Dm * 2);
    float*  alin = (float*) alloc((size_t)Mm * HNm * 4);
    float*  blin = (float*) alloc((size_t)Mm * HNm * 4);
    half_t* hsb  = (half_t*)alloc((size_t)Mm * HNm * 2);
    half_t* big0 = (half_t*)alloc((size_t)Mm * DFFm * 2);  // silu(t1) / gelu(g1)
    half_t* big1 = (half_t*)alloc((size_t)Mm * DFFm * 2);  // t2
    half_t *aWt[2], *iWt[2], *oWt[2], *w1t[2], *w2t[2], *w3t[2];
    for (int l = 0; l < 2; ++l) {
        aWt[l] = (half_t*)alloc((size_t)Dm * HNm * 2);
        iWt[l] = (half_t*)alloc((size_t)Dm * HNm * 2);
        oWt[l] = (half_t*)alloc((size_t)HNm * Dm * 2);
        w1t[l] = (half_t*)alloc((size_t)Dm * DFFm * 2);
        w2t[l] = (half_t*)alloc((size_t)DFFm * Dm * 2);
        w3t[l] = (half_t*)alloc((size_t)Dm * DFFm * 2);
    }
    half_t* h1t = (half_t*)alloc((size_t)Dm * 2 * Dm * 2);
    half_t* h2t = (half_t*)alloc((size_t)2 * Dm * Cm * 2);

    auto convtr = [&](const float* W, half_t* Wt, int K, int N) {
        const size_t tot = (size_t)K * N;
        convtr_k<<<(unsigned)((tot + 255) / 256), 256, 0, stream>>>(W, Wt, K, N);
    };
    for (int l = 0; l < 2; ++l) {
        convtr(alphaW + (size_t)l * Dm * HNm,  aWt[l], Dm,   HNm);
        convtr(inW    + (size_t)l * Dm * HNm,  iWt[l], Dm,   HNm);
        convtr(outW   + (size_t)l * HNm * Dm,  oWt[l], HNm,  Dm);
        convtr(w1W    + (size_t)l * Dm * DFFm, w1t[l], Dm,   DFFm);
        convtr(w2W    + (size_t)l * DFFm * Dm, w2t[l], DFFm, Dm);
        convtr(w3W    + (size_t)l * Dm * DFFm, w3t[l], Dm,   DFFm);
    }
    convtr(h1W, h1t, Dm, 2 * Dm);
    convtr(h2W, h2t, 2 * Dm, Cm);

    {
        const size_t tot = (size_t)Mm * Dm;
        embed_k<<<(unsigned)((tot + 255) / 256), 256, 0, stream>>>(x, emb, pos, h,
                                                                   Mm, Ssz, Dm);
    }

    auto gemm = [&](int EPI, const half_t* A, const half_t* Bt, const float* bias,
                    const float* res, const half_t* aux, float* Cf, half_t* Cb,
                    int Mx, int Nx, int Kx) {
        dim3 grid(Nx / BN, Mx / BM);
        switch (EPI) {
        case 0: gemm_f16_k<0><<<grid, 256, 0, stream>>>(A, Bt, bias, res, aux, Cf, Cb, Mx, Nx, Kx); break;
        case 1: gemm_f16_k<1><<<grid, 256, 0, stream>>>(A, Bt, bias, res, aux, Cf, Cb, Mx, Nx, Kx); break;
        case 2: gemm_f16_k<2><<<grid, 256, 0, stream>>>(A, Bt, bias, res, aux, Cf, Cb, Mx, Nx, Kx); break;
        case 3: gemm_f16_k<3><<<grid, 256, 0, stream>>>(A, Bt, bias, res, aux, Cf, Cb, Mx, Nx, Kx); break;
        default: gemm_f16_k<4><<<grid, 256, 0, stream>>>(A, Bt, bias, res, aux, Cf, Cb, Mx, Nx, Kx); break;
        }
    };

    for (int l = 0; l < 2; ++l) {
        // ---- SSM block
        ln_k<<<Mm, 256, 0, stream>>>(h, ssmG + (size_t)l * Dm, ssmBe + (size_t)l * Dm, xnb, Dm);
        gemm(0, xnb, aWt[l], alphaB + (size_t)l * HNm, nullptr, nullptr, alin, nullptr, Mm, HNm, Dm);
        gemm(0, xnb, iWt[l], inB    + (size_t)l * HNm, nullptr, nullptr, blin, nullptr, Mm, HNm, Dm);
        scan_k<<<2, 256, 0, stream>>>(alin, blin, hsb, Bsz, Ssz, HNm);
        gemm(1, hsb, oWt[l], outB + (size_t)l * Dm, h, nullptr, h, nullptr, Mm, Dm, HNm);
        // ---- SwiGLU block
        ln_k<<<Mm, 256, 0, stream>>>(h, ffG + (size_t)l * Dm, ffBe + (size_t)l * Dm, xnb, Dm);
        gemm(2, xnb, w1t[l], w1B + (size_t)l * DFFm, nullptr, nullptr, nullptr, big0, Mm, DFFm, Dm);
        gemm(3, xnb, w3t[l], w3B + (size_t)l * DFFm, nullptr, big0, nullptr, big1, Mm, DFFm, Dm);
        gemm(1, big1, w2t[l], w2B + (size_t)l * Dm, h, nullptr, h, nullptr, Mm, Dm, DFFm);
    }

    // ---- head
    ln_k<<<Mm, 256, 0, stream>>>(h, onG, onBe, xnb, Dm);
    gemm(4, xnb, h1t, h1B, nullptr, nullptr, nullptr, big0, Mm, 2 * Dm, Dm);
    gemm(0, big0, h2t, h2B, nullptr, nullptr, outp, nullptr, Mm, Cm, 2 * Dm);
}
